// SASRec_10634339025420
// MI455X (gfx1250) — matrix operand order repaired and checked by reading, not compiled, not run on hardware
//
#include <hip/hip_runtime.h>

// ---------------------------------------------------------------------------
// Problem constants (match reference)
// ---------------------------------------------------------------------------
#define B_      8
#define S_      2048
#define NBS     16384          // B_*S_
#define DIMC    512            // EMB 448 + FEAT 64
#define EMBD    448
#define FEATD   64
#define MSZ     (512 * 512)    // one weight matrix, elements

// ---------------------------------------------------------------------------
// WMMA types / helpers (CDNA5 gfx1250, wave32)
// ---------------------------------------------------------------------------
typedef __attribute__((ext_vector_type(16))) __bf16 v16bf;
typedef __attribute__((ext_vector_type(8)))  __bf16 v8bf;
typedef __attribute__((ext_vector_type(8)))  float  v8f;

__device__ __forceinline__ v8bf ld8(const __bf16* p) {
    return *reinterpret_cast<const v8bf*>(p);
}
__device__ __forceinline__ v16bf cat16(v8bf lo, v8bf hi) {
    return __builtin_shufflevector(lo, hi, 0,1,2,3,4,5,6,7,8,9,10,11,12,13,14,15);
}

// A fragment: 16x32 bf16 (MxK). Lane L holds row M=L&15; half=L>>4 selects
// K-interleave: elems 0..7 -> K = kc + half*8 + i ; elems 8..15 -> K = kc+16+half*8+i
__device__ __forceinline__ v16bf load_a_frag(const __bf16* A, int row0, int ldk,
                                             int kc, int lane) {
    const int row  = lane & 15;
    const int half = lane >> 4;
    const __bf16* p = A + (size_t)(row0 + row) * ldk + kc + half * 8;
    return cat16(ld8(p), ld8(p + 16));
}

// B fragment: 32x16 bf16 (KxN), sourced from row-major Wn[k] (i.e. B[k][n]=W row n).
// Lane L holds column N=L&15; elems 0..7 -> K = kc + (L>>4)*8 + i ; elems 8..15 -> K = kc+16+(L>>4)*8+i
__device__ __forceinline__ v16bf load_b_frag(const __bf16* W, int nbase, int ldk,
                                             int kc, int lane) {
    const int col  = lane & 15;
    const int half = lane >> 4;
    const __bf16* p = W + (size_t)(nbase + col) * ldk + kc + half * 8;
    return cat16(ld8(p), ld8(p + 16));
}

__device__ __forceinline__ v8f wmma_bf16(v16bf a, v16bf b, v8f c) {
    return __builtin_amdgcn_wmma_f32_16x16x32_bf16(
        /*neg_a=*/false, a, /*neg_b=*/false, b,
        /*c_mod=*/(short)0, c, /*reuse_a=*/false, /*reuse_b=*/false);
}

// ---------------------------------------------------------------------------
// fp32 -> bf16 conversion (weights)
// ---------------------------------------------------------------------------
__global__ void f2bf_kernel(const float* __restrict__ in,
                            unsigned short* __restrict__ outRaw, int n) {
    int i = blockIdx.x * 256 + threadIdx.x;
    if (i < n) {
        __bf16* out = reinterpret_cast<__bf16*>(outRaw);
        out[i] = (__bf16)in[i];
    }
}

// ---------------------------------------------------------------------------
// Embedding gather + feature concat -> bf16 x  [NBS][512]
// ---------------------------------------------------------------------------
__global__ __launch_bounds__(128) void gather_concat_kernel(
    const int* __restrict__ seq, const float* __restrict__ feats,
    const float* __restrict__ emb, unsigned short* __restrict__ xbRaw) {
    const int row  = blockIdx.x;          // b*S + s
    const int item = seq[row];
    const float* e = emb   + (size_t)item * EMBD;
    const float* f = feats + (size_t)row  * FEATD;
    __bf16* o = reinterpret_cast<__bf16*>(xbRaw) + (size_t)row * DIMC;
#pragma unroll
    for (int j = 0; j < 4; ++j) {
        int d = threadIdx.x * 4 + j;
        float v = (d < EMBD) ? e[d] : f[d - EMBD];
        o[d] = (__bf16)v;
    }
}

// ---------------------------------------------------------------------------
// GEMM: C[M=16384][512] = A[M][512] @ W[512][512]^T + bias
//   grid = M/32 blocks, 256 threads = 8 waves.
//   32x512 bf16 A panel staged in LDS (32 KiB) once per block, then each wave
//   computes 2 M-tiles x 4 N-tiles (8 WMMA accumulators, cols [w*64, w*64+64)).
//   RELU:   apply max(0, .)        (FFN1)
//   F32OUT: also write fp32 result (FFN2 -> final x for output extraction)
//   TRANSV: write bf16 transposed as [B][512][2048]  (V for attention),
//           one packed 16-B store per lane per tile.
// ---------------------------------------------------------------------------
template <bool RELU, bool F32OUT, bool TRANSV>
__global__ __launch_bounds__(256) void gemm_wmma_kernel(
    const unsigned short* __restrict__ ARaw,
    const unsigned short* __restrict__ WRaw,
    const float* __restrict__ bias,
    unsigned short* __restrict__ outRaw,
    float* __restrict__ outF) {
    __shared__ __attribute__((aligned(16))) __bf16 sA[32 * DIMC];   // 32 KiB

    const __bf16* A = reinterpret_cast<const __bf16*>(ARaw);
    const __bf16* W = reinterpret_cast<const __bf16*>(WRaw);
    __bf16* outB    = reinterpret_cast<__bf16*>(outRaw);

    const int lane  = threadIdx.x & 31;
    const int wave  = threadIdx.x >> 5;
    const int mbase = blockIdx.x * 32;
    const int nbase = wave * 64;

    // stage A rows [mbase, mbase+32) -> LDS (contiguous 32 KiB, coalesced b128)
    {
        const uint4* src = reinterpret_cast<const uint4*>(A + (size_t)mbase * DIMC);
        uint4* dst = reinterpret_cast<uint4*>(sA);
#pragma unroll
        for (int i = 0; i < 8; ++i)
            dst[i * 256 + threadIdx.x] = src[i * 256 + threadIdx.x];
    }
    __syncthreads();

    v8f acc[2][4] = {};
    const int half = lane >> 4;
    for (int kc = 0; kc < DIMC; kc += 32) {
        v16bf bfr[4];
#pragma unroll
        for (int t = 0; t < 4; ++t)
            bfr[t] = load_b_frag(W, nbase + t * 16, DIMC, kc, lane);
        __builtin_prefetch(W + (size_t)(nbase + (lane & 15)) * DIMC + kc + 64, 0, 0);
#pragma unroll
        for (int mt = 0; mt < 2; ++mt) {
            const __bf16* sp = &sA[(size_t)(mt * 16 + (lane & 15)) * DIMC + kc + half * 8];
            v16bf a = cat16(*reinterpret_cast<const v8bf*>(sp),
                            *reinterpret_cast<const v8bf*>(sp + 16));
#pragma unroll
            for (int t = 0; t < 4; ++t)
                acc[mt][t] = wmma_bf16(a, bfr[t], acc[mt][t]);
        }
    }

    const int col = lane & 15;
#pragma unroll
    for (int mt = 0; mt < 2; ++mt) {
#pragma unroll
        for (int t = 0; t < 4; ++t) {
            const int n  = nbase + t * 16 + col;
            const float bv = bias[n];
            if (TRANSV) {
                v8bf pk;
#pragma unroll
                for (int r = 0; r < 8; ++r) pk[r] = (__bf16)(acc[mt][t][r] + bv);
                const int m0 = mbase + mt * 16 + 8 * half;   // 8 consecutive keys
                const int bb = m0 >> 11;                     // m0 / 2048
                const int ss = m0 & 2047;
                *reinterpret_cast<v8bf*>(outB + ((size_t)bb * DIMC + n) * S_ + ss) = pk;
            } else {
#pragma unroll
                for (int r = 0; r < 8; ++r) {
                    const int m = mbase + mt * 16 + r + 8 * half;
                    float v = acc[mt][t][r] + bv;
                    if (RELU) v = fmaxf(v, 0.0f);
                    if (F32OUT) outF[(size_t)m * DIMC + n] = v;
                    outB[(size_t)m * DIMC + n] = (__bf16)v;
                }
            }
        }
    }
}

// ---------------------------------------------------------------------------
// Causal (unsoftmaxed) attention:  H = tril(Q K^T * scale) @ V
//   grid = B * S/16 blocks; block owns one 16-row q tile; 8 waves.
//   Q tile (16x512 bf16, 16 KiB) staged in LDS once. Per 128-key block:
//   wave w computes 16x16 score tile for keys [kb0+w*16, +16) (16 bf16 WMMAs
//   over D=512, A from LDS), scales + tril-masks, stages bf16 scores in LDS;
//   then every wave accumulates its 4 d-tiles (64 cols) with
//   scores(A, LDS) x V^T(B, contiguous global) WMMAs.
// ---------------------------------------------------------------------------
__global__ __launch_bounds__(256) void attn_wmma_kernel(
    const unsigned short* __restrict__ QRaw,
    const unsigned short* __restrict__ KRaw,
    const unsigned short* __restrict__ VTRaw,
    float* __restrict__ H) {
    __shared__ __attribute__((aligned(16))) __bf16 sQ[16 * DIMC];   // 16 KiB
    __shared__ __attribute__((aligned(32))) __bf16 s_sc[16 * 128];  //  4 KiB

    const __bf16* Q  = reinterpret_cast<const __bf16*>(QRaw);
    const __bf16* K  = reinterpret_cast<const __bf16*>(KRaw);
    const __bf16* VT = reinterpret_cast<const __bf16*>(VTRaw);

    const int lane  = threadIdx.x & 31;
    const int wave  = threadIdx.x >> 5;
    const int bidx  = blockIdx.x >> 7;     // / (S/16)
    const int qt    = blockIdx.x & 127;
    const int qbase = qt * 16;

    const __bf16* Qb = Q  + (size_t)bidx * S_ * DIMC;
    const __bf16* Kb = K  + (size_t)bidx * S_ * DIMC;
    const __bf16* Vb = VT + (size_t)bidx * DIMC * S_;

    // stage Q tile -> LDS (16 KiB contiguous, coalesced b128)
    {
        const uint4* src = reinterpret_cast<const uint4*>(Qb + (size_t)qbase * DIMC);
        uint4* dst = reinterpret_cast<uint4*>(sQ);
#pragma unroll
        for (int i = 0; i < 4; ++i)
            dst[i * 256 + threadIdx.x] = src[i * 256 + threadIdx.x];
    }

    const int col  = lane & 15;
    const int half = lane >> 4;
    const float scale = 0.044194173824159216f;   // 1/sqrt(512)

    v8f acc[4] = {};
    const int kend = qbase + 16;
    for (int kb0 = 0; kb0 < kend; kb0 += 128) {
        __syncthreads();                         // also covers sQ staging (1st iter)
        const int keyt = kb0 + wave * 16;        // wave-uniform
        if (keyt <= qbase) {                     // tile has at least one valid key
            v8f sc = {};
            for (int kc = 0; kc < DIMC; kc += 32) {
                const __bf16* sp = &sQ[(size_t)col * DIMC + kc + half * 8];
                v16bf a = cat16(*reinterpret_cast<const v8bf*>(sp),
                                *reinterpret_cast<const v8bf*>(sp + 16));
                v16bf b = load_b_frag(Kb, keyt, DIMC, kc, lane);
                sc = wmma_bf16(a, b, sc);
            }
            const int key = keyt + col;
#pragma unroll
            for (int r = 0; r < 8; ++r) {
                const int qrow = qbase + r + 8 * half;
                const float v = (key <= qrow) ? sc[r] * scale : 0.0f;
                s_sc[(r + 8 * half) * 128 + wave * 16 + col] = (__bf16)v;
            }
        } else {
#pragma unroll
            for (int r = 0; r < 8; ++r)
                s_sc[(r + 8 * half) * 128 + wave * 16 + col] = (__bf16)0.0f;
        }
        __syncthreads();
#pragma unroll
        for (int kc2 = 0; kc2 < 4; ++kc2) {
            const __bf16* sp = &s_sc[(lane & 15) * 128 + kc2 * 32 + half * 8];
            v16bf a = cat16(*reinterpret_cast<const v8bf*>(sp),
                            *reinterpret_cast<const v8bf*>(sp + 16));
#pragma unroll
            for (int t = 0; t < 4; ++t) {
                const int d0 = wave * 64 + t * 16;
                v16bf b = load_b_frag(Vb, d0, S_, kb0 + kc2 * 32, lane);
                acc[t] = wmma_bf16(a, b, acc[t]);
            }
        }
    }

    float* Hb = H + ((size_t)bidx * S_ + qbase) * DIMC;
#pragma unroll
    for (int t = 0; t < 4; ++t) {
        const int d = wave * 64 + t * 16 + col;
#pragma unroll
        for (int r = 0; r < 8; ++r)
            Hb[(size_t)(r + 8 * half) * DIMC + d] = acc[t][r];
    }
}

// ---------------------------------------------------------------------------
// Per-(b,d) sequence norm over S axis (ddof=1, reference's double 1e-5),
// fp32 in -> normalized bf16 out. Coalesced: block = (batch, 32 d-cols);
// lanes span consecutive d, 8 s-groups reduced via LDS. Purely BW-bound.
// ---------------------------------------------------------------------------
__global__ __launch_bounds__(256) void seq_norm_kernel(
    const float* __restrict__ H, unsigned short* __restrict__ HBRaw) {
    __shared__ float rs[8][32], rq[8][32], bm[32], bi[32];
    const int lane = threadIdx.x & 31;     // d offset within group of 32
    const int g    = threadIdx.x >> 5;     // s group (0..7)
    const int b    = blockIdx.x >> 4;
    const int d    = ((blockIdx.x & 15) << 5) + lane;

    const float* p = H + (size_t)b * S_ * DIMC + d;
    float s = 0.0f, q = 0.0f;
    for (int i = g; i < S_; i += 8) {
        const float v = p[(size_t)i * DIMC];
        s += v; q += v * v;
    }
    rs[g][lane] = s; rq[g][lane] = q;
    __syncthreads();
    if (g == 0) {
        float S = 0.0f, Q = 0.0f;
#pragma unroll
        for (int j = 0; j < 8; ++j) { S += rs[j][lane]; Q += rq[j][lane]; }
        const float mean = S * (1.0f / (float)S_);
        const float var  = fmaxf((Q - S * mean) / (float)(S_ - 1), 0.0f);
        bm[lane] = mean;
        bi[lane] = 1.0f / (sqrtf(var) + 1e-5f + 1e-5f);
    }
    __syncthreads();
    const float mean = bm[lane], inv = bi[lane];
    __bf16* o = reinterpret_cast<__bf16*>(HBRaw) + (size_t)b * S_ * DIMC + d;
    for (int i = g; i < S_; i += 8) {
        const float v = p[(size_t)i * DIMC];
        o[(size_t)i * DIMC] = (__bf16)((v - mean) * inv);
    }
}

// ---------------------------------------------------------------------------
// Output: d_out[b][d] = xf[b][S-1][d]
// ---------------------------------------------------------------------------
__global__ void extract_last_kernel(const float* __restrict__ XF,
                                    float* __restrict__ out) {
    const int i = blockIdx.x * 256 + threadIdx.x;   // 0 .. 4095
    if (i < B_ * DIMC) {
        const int b = i >> 9, d = i & 511;
        out[i] = XF[((size_t)b * S_ + (S_ - 1)) * DIMC + d];
    }
}

// ---------------------------------------------------------------------------
// Host orchestration
// ---------------------------------------------------------------------------
extern "C" void kernel_launch(void* const* d_in, const int* in_sizes, int n_in,
                              void* d_out, int out_size, void* d_ws, size_t ws_size,
                              hipStream_t stream) {
    (void)in_sizes; (void)n_in; (void)out_size; (void)ws_size;

    const int*   seq   = (const int*)  d_in[0];
    const float* feats = (const float*)d_in[1];
    const float* emb   = (const float*)d_in[2];
    const float* Wq    = (const float*)d_in[3];
    const float* bq    = (const float*)d_in[4];
    const float* Wk    = (const float*)d_in[5];
    const float* bk    = (const float*)d_in[6];
    const float* Wv    = (const float*)d_in[7];
    const float* bv    = (const float*)d_in[8];
    const float* W1    = (const float*)d_in[9];
    const float* b1    = (const float*)d_in[10];
    const float* W2    = (const float*)d_in[11];
    const float* b2    = (const float*)d_in[12];
    float* out = (float*)d_out;

    // workspace layout
    char* ws = (char*)d_ws;
    const size_t SZ_BF  = (size_t)NBS * DIMC * 2;   // 16 MiB
    const size_t SZ_F32 = (size_t)NBS * DIMC * 4;   // 32 MiB
    unsigned short* xb = (unsigned short*)(ws + 0 * SZ_BF);
    unsigned short* qb = (unsigned short*)(ws + 1 * SZ_BF);
    unsigned short* kb = (unsigned short*)(ws + 2 * SZ_BF);
    unsigned short* vt = (unsigned short*)(ws + 3 * SZ_BF);
    unsigned short* hb = (unsigned short*)(ws + 4 * SZ_BF);
    unsigned short* t1 = (unsigned short*)(ws + 5 * SZ_BF);
    float*          hf = (float*)        (ws + 6 * SZ_BF);
    float*          xf = (float*)        (ws + 6 * SZ_BF + SZ_F32);
    unsigned short* wb = (unsigned short*)(ws + 6 * SZ_BF + 2 * SZ_F32);
    // bf16 weight regions: [Wq(2), Wk(2), Wv(2), W1(2), W2(2)] x 512x512
    unsigned short* wqB = wb + 0 * 2 * MSZ;
    unsigned short* wkB = wb + 1 * 2 * MSZ;
    unsigned short* wvB = wb + 2 * 2 * MSZ;
    unsigned short* w1B = wb + 3 * 2 * MSZ;
    unsigned short* w2B = wb + 4 * 2 * MSZ;

    const int cn = 2 * MSZ;                 // elements per weight tensor
    const dim3 cgrid((cn + 255) / 256);
    f2bf_kernel<<<cgrid, 256, 0, stream>>>(Wq, wqB, cn);
    f2bf_kernel<<<cgrid, 256, 0, stream>>>(Wk, wkB, cn);
    f2bf_kernel<<<cgrid, 256, 0, stream>>>(Wv, wvB, cn);
    f2bf_kernel<<<cgrid, 256, 0, stream>>>(W1, w1B, cn);
    f2bf_kernel<<<cgrid, 256, 0, stream>>>(W2, w2B, cn);

    gather_concat_kernel<<<NBS, 128, 0, stream>>>(seq, feats, emb, xb);

    const dim3 ggrid(NBS / 32);   // 512 blocks (32 M-rows each)
    const dim3 agrid(B_ * S_ / 16);
    for (int l = 0; l < 2; ++l) {
        const unsigned short* wq_l = wqB + (size_t)l * MSZ;
        const unsigned short* wk_l = wkB + (size_t)l * MSZ;
        const unsigned short* wv_l = wvB + (size_t)l * MSZ;
        const unsigned short* w1_l = w1B + (size_t)l * MSZ;
        const unsigned short* w2_l = w2B + (size_t)l * MSZ;
        const float* bq_l = bq + (size_t)l * DIMC;
        const float* bk_l = bk + (size_t)l * DIMC;
        const float* bv_l = bv + (size_t)l * DIMC;
        const float* b1_l = b1 + (size_t)l * DIMC;
        const float* b2_l = b2 + (size_t)l * DIMC;

        // q = x Wq^T + bq  (bf16)
        gemm_wmma_kernel<false, false, false><<<ggrid, 256, 0, stream>>>(
            xb, wq_l, bq_l, qb, (float*)nullptr);
        // k = x Wk^T + bk  (bf16)
        gemm_wmma_kernel<false, false, false><<<ggrid, 256, 0, stream>>>(
            xb, wk_l, bk_l, kb, (float*)nullptr);
        // v = x Wv^T + bv  (bf16, transposed [B][512][2048])
        gemm_wmma_kernel<false, false, true><<<ggrid, 256, 0, stream>>>(
            xb, wv_l, bv_l, vt, (float*)nullptr);

        // h = tril(q k^T * scale) v   (fp32)
        attn_wmma_kernel<<<agrid, 256, 0, stream>>>(qb, kb, vt, hf);

        // sequence norm -> bf16
        seq_norm_kernel<<<B_ * 16, 256, 0, stream>>>(hf, hb);

        // FFN1: relu(h W1^T + b1) -> bf16
        gemm_wmma_kernel<true, false, false><<<ggrid, 256, 0, stream>>>(
            hb, w1_l, b1_l, t1, (float*)nullptr);
        // FFN2: t1 W2^T + b2 -> bf16 (next x, overwrites xb) + fp32 (xf)
        gemm_wmma_kernel<false, true, false><<<ggrid, 256, 0, stream>>>(
            t1, w2_l, b2_l, xb, xf);
    }

    extract_last_kernel<<<(B_ * DIMC + 255) / 256, 256, 0, stream>>>(xf, out);
}
